// QWenAttention_59219009077592
// MI455X (gfx1250) — compile-verified, matrix operand order checked by reading, not run on hardware
//
#include <hip/hip_runtime.h>
#include <hip/hip_bf16.h>

// ---------------------------------------------------------------------------
// Problem constants (QWen attention reference)
// ---------------------------------------------------------------------------
constexpr int B  = 2;
constexpr int S  = 2048;
constexpr int D  = 4096;
constexpr int H  = 32;
constexpr int Dh = 128;            // D / H
constexpr float SCALE = 0.08838834764831845f;   // Dh^-0.5
constexpr float ROPE_BASE = 10000.0f;

typedef __bf16 bf16x8  __attribute__((ext_vector_type(8)));
typedef __bf16 bf16x16 __attribute__((ext_vector_type(16)));
typedef float  f32x8   __attribute__((ext_vector_type(8)));
typedef int    v4i     __attribute__((ext_vector_type(4)));

// ---------------------------------------------------------------------------
// CDNA5 async global->LDS copy path (ASYNCcnt), with sync fallback
// ---------------------------------------------------------------------------
#if __has_builtin(__builtin_amdgcn_global_load_async_to_lds_b128)
#define HAVE_ASYNC_LDS 1
#else
#define HAVE_ASYNC_LDS 0
#endif

__device__ __forceinline__ void copy16_to_lds(const void* g, void* l) {
#if HAVE_ASYNC_LDS
    // Builtin signature: (int4 addrspace(1)*, int4 addrspace(3)*, imm offset, imm cpol).
    // AS3 pointers are 32-bit; low 32 bits of a generic LDS pointer are the
    // LDS byte offset.  AS1 is the plain 64-bit global address.
    __builtin_amdgcn_global_load_async_to_lds_b128(
        (__attribute__((address_space(1))) v4i*)(uintptr_t)g,
        (__attribute__((address_space(3))) v4i*)(unsigned)(uintptr_t)l,
        0, 0);
#else
    *reinterpret_cast<uint4*>(l) = *reinterpret_cast<const uint4*>(g);
#endif
}

__device__ __forceinline__ void async_wait_all() {
#if HAVE_ASYNC_LDS
#if __has_builtin(__builtin_amdgcn_s_wait_asynccnt)
    __builtin_amdgcn_s_wait_asynccnt(0);
#else
    asm volatile("s_wait_asynccnt 0x0" ::: "memory");
#endif
#endif
}

// ---------------------------------------------------------------------------
// Helpers
// ---------------------------------------------------------------------------
__device__ __forceinline__ __bf16 f2bf(float f) {
    union { float f; unsigned u; } a; a.f = f;
    unsigned r = a.u + 0x7fffu + ((a.u >> 16) & 1u);   // RNE
    union { unsigned short s; __bf16 b; } o;
    o.s = (unsigned short)(r >> 16);
    return o.b;
}
__device__ __forceinline__ float bf2f(__bf16 b) {
    union { unsigned short s; __bf16 b; } i; i.b = b;
    union { unsigned u; float f; } o;
    o.u = ((unsigned)i.s) << 16;
    return o.f;
}

// Load a 16x32 bf16 A/B fragment from a row-major (row, K-contiguous) tile.
// CDNA5 16-bit layout: lane = row (lane&15); element j<8 -> K = base+j,
// j>=8 -> K = base+16+(j-8), base = (lane>>4)*8.  -> two 16-byte loads/lane.
__device__ __forceinline__ bf16x16 frag_ld(const __bf16* p0, int ld) {
    const int lane = threadIdx.x & 31;
    const __bf16* p = p0 + (lane & 15) * ld + (lane >> 4) * 8;
    bf16x8 lo = *reinterpret_cast<const bf16x8*>(p);
    bf16x8 hi = *reinterpret_cast<const bf16x8*>(p + 16);
    bf16x16 r;
#pragma unroll
    for (int i = 0; i < 8; ++i) { r[i] = lo[i]; r[i + 8] = hi[i]; }
    return r;
}

// ---------------------------------------------------------------------------
// Kernel 1: f32 -> bf16 elementwise
// ---------------------------------------------------------------------------
__global__ void cvt_bf16_kernel(const float* __restrict__ x,
                                __bf16* __restrict__ y, size_t n) {
    size_t i = (size_t)blockIdx.x * blockDim.x + threadIdx.x;
    if (i < n) y[i] = f2bf(x[i]);
}

// ---------------------------------------------------------------------------
// Kernel 2: transpose (K x N f32) -> (N x K bf16) so WMMA B-fragments read
// contiguous K.
// ---------------------------------------------------------------------------
__global__ __launch_bounds__(256)
void transpose_cvt_kernel(const float* __restrict__ W, __bf16* __restrict__ Wt,
                          int K, int N) {
    __shared__ float tile[32][33];
    const int n0 = blockIdx.x * 32, k0 = blockIdx.y * 32;
    const int tx = threadIdx.x, ty = threadIdx.y;      // 32 x 8
#pragma unroll
    for (int i = 0; i < 32; i += 8)
        tile[ty + i][tx] = W[(size_t)(k0 + ty + i) * N + n0 + tx];
    __syncthreads();
#pragma unroll
    for (int i = 0; i < 32; i += 8)
        Wt[(size_t)(n0 + ty + i) * K + k0 + tx] = f2bf(tile[tx][ty + i]);
}

// ---------------------------------------------------------------------------
// Kernel 3: double-buffered WMMA GEMM.  C[M,N] = A[M,K] * Bt[N,K]^T (+bias)
// Block 256 threads (8 waves), tile 128x128, K-step 32, async global->LDS.
// Wave (wm in 0..3, wn in 0..1) owns a 32x64 sub-tile -> 2x4 wmma accums.
// ---------------------------------------------------------------------------
template <bool HAS_BIAS, bool OUT_BF16>
__global__ __launch_bounds__(256)
void gemm_wmma_kernel(const __bf16* __restrict__ A, const __bf16* __restrict__ Bt,
                      const float* __restrict__ bias, void* __restrict__ out,
                      int M, int N, int K) {
    constexpr int LD = 40;                       // 32 + 8 pad, keeps 16B align
    __shared__ __bf16 As[2][128 * LD];
    __shared__ __bf16 Bs[2][128 * LD];
    const int t = threadIdx.x;
    const int lane = t & 31, wave = t >> 5;
    const int wm = wave & 3, wn = wave >> 2;
    const int m0 = blockIdx.y * 128, n0 = blockIdx.x * 128;

    f32x8 acc[2][4] = {};

    // issue one 128x32 A slab + 128x32 B slab into stage `st`
    auto issue_tiles = [&](int k0, int st) {
#pragma unroll
        for (int i = 0; i < 2; ++i) {
            int c = t + i * 256;                 // 512 16B chunks per operand
            int row = c >> 2, c16 = c & 3;
            copy16_to_lds(&A[(size_t)(m0 + row) * K + k0 + c16 * 8],
                          &As[st][row * LD + c16 * 8]);
            copy16_to_lds(&Bt[(size_t)(n0 + row) * K + k0 + c16 * 8],
                          &Bs[st][row * LD + c16 * 8]);
        }
    };

    const int nsteps = K / 32;
    issue_tiles(0, 0);

    for (int s = 0; s < nsteps; ++s) {
        async_wait_all();
        __syncthreads();                         // stage s%2 ready for all waves
        if (s + 1 < nsteps) issue_tiles((s + 1) * 32, (s + 1) & 1);

        const __bf16* Ab = As[s & 1];
        const __bf16* Bb = Bs[s & 1];
        bf16x16 af[2], bfrag[4];
#pragma unroll
        for (int i = 0; i < 2; ++i)
            af[i] = frag_ld(&Ab[(wm * 32 + i * 16) * LD], LD);
#pragma unroll
        for (int j = 0; j < 4; ++j)
            bfrag[j] = frag_ld(&Bb[(wn * 64 + j * 16) * LD], LD);
#pragma unroll
        for (int i = 0; i < 2; ++i)
#pragma unroll
            for (int j = 0; j < 4; ++j)
                acc[i][j] = __builtin_amdgcn_wmma_f32_16x16x32_bf16(
                    false, af[i], false, bfrag[j], (short)0, acc[i][j], false, false);
        __syncthreads();                         // protect stage s%2 before reuse
    }

    // Epilogue: C layout -> lane holds col n=lane&15, rows r + 8*(lane>>4)
    const int half = lane >> 4, nn = lane & 15;
#pragma unroll
    for (int i = 0; i < 2; ++i)
#pragma unroll
        for (int j = 0; j < 4; ++j) {
            int cg = n0 + wn * 64 + j * 16 + nn;
            float bv = HAS_BIAS ? bias[cg] : 0.0f;
#pragma unroll
            for (int r = 0; r < 8; ++r) {
                int rg = m0 + wm * 32 + i * 16 + r + 8 * half;
                float v = acc[i][j][r] + bv;
                if (OUT_BF16)
                    ((__bf16*)out)[(size_t)rg * N + cg] = f2bf(v);
                else
                    ((float*)out)[(size_t)rg * N + cg] = v;
            }
        }
}

// ---------------------------------------------------------------------------
// Kernel 4: RoPE + repack.
//  qkv [B,S,3D] bf16  ->  Q [B,H,S,Dh] (pre-scaled by Dh^-0.5),
//                         K [B,H,S,Dh],  Vt [B,H,Dh,S]
// ---------------------------------------------------------------------------
__global__ __launch_bounds__(256)
void rope_pack_kernel(const __bf16* __restrict__ qkv, const int* __restrict__ pos,
                      __bf16* __restrict__ Q, __bf16* __restrict__ Kd,
                      __bf16* __restrict__ Vt) {
    const int bs = blockIdx.x;                  // b*S + s
    const int b = bs / S, s = bs % S;
    const float p = (float)pos[bs];
    const __bf16* base = qkv + (size_t)bs * (3 * D);

    for (int e = threadIdx.x; e < D; e += blockDim.x) {
        const int h = e >> 7, d = e & 127;
        const int f = d & 63;
        const float ang = p * __powf(ROPE_BASE, -(float)f * (1.0f / 64.0f));
        float sn, cs;
        __sincosf(ang, &sn, &cs);
        const int ep = (d < 64) ? e + 64 : e - 64;

        float x  = bf2f(base[e]);
        float xp = bf2f(base[ep]);
        float q  = (d < 64) ? (x * cs - xp * sn) : (x * cs + xp * sn);

        float y  = bf2f(base[D + e]);
        float yp = bf2f(base[D + ep]);
        float k  = (d < 64) ? (y * cs - yp * sn) : (y * cs + yp * sn);

        const size_t qi = (((size_t)b * H + h) * S + s) * Dh + d;
        Q[qi]  = f2bf(q * SCALE);
        Kd[qi] = f2bf(k);

        float v = bf2f(base[2 * D + e]);
        Vt[(((size_t)b * H + h) * Dh + d) * S + s] = f2bf(v);
    }
}

// ---------------------------------------------------------------------------
// Kernel 5: causal flash attention, double-buffered K/V tiles.
// Block = (qt, h, b): 64 q rows, 4 waves x 16 rows, stream 64-key tiles.
// ---------------------------------------------------------------------------
__global__ __launch_bounds__(128)
void flash_attn_kernel(const __bf16* __restrict__ Q, const __bf16* __restrict__ Kd,
                       const __bf16* __restrict__ Vt, __bf16* __restrict__ ctx) {
    constexpr int LDK = 136;                    // 128 + 8 pad
    constexpr int LDV = 72;                     // 64 + 8 pad
    constexpr int LDP = 72;
    __shared__ __bf16 Ks[2][64 * LDK];
    __shared__ __bf16 Vs[2][128 * LDV];
    __shared__ __bf16 Ps[4 * 16 * LDP];

    const int t = threadIdx.x, lane = t & 31, wave = t >> 5;
    const int qt = blockIdx.x, h = blockIdx.y, b = blockIdx.z;
    const size_t headQK = ((size_t)b * H + h) * S;   // rows of Q/K
    const size_t headV  = ((size_t)b * H + h) * Dh;  // rows of Vt
    const int q0 = qt * 64;
    const int qw = q0 + wave * 16;
    const int half = lane >> 4, nn = lane & 15;

    // Q fragments for this wave's 16 rows (kept in registers all loop long)
    bf16x16 qf[4];
#pragma unroll
    for (int f = 0; f < 4; ++f)
        qf[f] = frag_ld(Q + (headQK + qw) * Dh + f * 32, Dh);

    f32x8 oacc[8] = {};
    float mrun[8], lrun[8];
#pragma unroll
    for (int r = 0; r < 8; ++r) { mrun[r] = -1e30f; lrun[r] = 0.0f; }

    __bf16* Pw = Ps + wave * 16 * LDP;

    // issue K (64x128) and Vt (128x64) tiles for key-block kt into stage st
    auto issue_tiles = [&](int kt, int st) {
        const int k0 = kt * 64;
#pragma unroll
        for (int i = 0; i < 8; ++i) {
            int c = t + i * 128;
            int row = c >> 4, c16 = c & 15;
            copy16_to_lds(&Kd[(headQK + k0 + row) * Dh + c16 * 8],
                          &Ks[st][row * LDK + c16 * 8]);
        }
#pragma unroll
        for (int i = 0; i < 8; ++i) {
            int c = t + i * 128;
            int row = c >> 3, c16 = c & 7;
            copy16_to_lds(&Vt[(headV + row) * S + k0 + c16 * 8],
                          &Vs[st][row * LDV + c16 * 8]);
        }
    };

    issue_tiles(0, 0);

    for (int kt = 0; kt <= qt; ++kt) {
        async_wait_all();
        __syncthreads();                        // stage kt&1 ready
        if (kt < qt) issue_tiles(kt + 1, (kt + 1) & 1);

        const __bf16* Kb = Ks[kt & 1];
        const __bf16* Vb = Vs[kt & 1];

        // S = Q * K^T  (16x64 per wave, 16 WMMAs)
        f32x8 sacc[4] = {};
#pragma unroll
        for (int nt = 0; nt < 4; ++nt)
#pragma unroll
            for (int f = 0; f < 4; ++f)
                sacc[nt] = __builtin_amdgcn_wmma_f32_16x16x32_bf16(
                    false, qf[f], false,
                    frag_ld(&Kb[(nt * 16) * LDK + f * 32], LDK),
                    (short)0, sacc[nt], false, false);

        // causal mask (only the diagonal tile needs it)
        if (kt == qt) {
#pragma unroll
            for (int nt = 0; nt < 4; ++nt)
#pragma unroll
                for (int r = 0; r < 8; ++r) {
                    int qrow = wave * 16 + r + 8 * half;
                    int kcol = nt * 16 + nn;
                    if (kcol > qrow) sacc[nt][r] = -1e30f;
                }
        }

        // online softmax: row r (+8*half) lives on 16 lanes at vgpr index r
#pragma unroll
        for (int r = 0; r < 8; ++r) {
            float mx = sacc[0][r];
#pragma unroll
            for (int nt = 1; nt < 4; ++nt) mx = fmaxf(mx, sacc[nt][r]);
#pragma unroll
            for (int off = 1; off < 16; off <<= 1)
                mx = fmaxf(mx, __shfl_xor(mx, off));
            float newm = fmaxf(mrun[r], mx);
            float scl = __expf(mrun[r] - newm);
            mrun[r] = newm;
            float sum = 0.0f;
#pragma unroll
            for (int nt = 0; nt < 4; ++nt) {
                float pv = __expf(sacc[nt][r] - newm);
                sacc[nt][r] = pv;
                sum += pv;
            }
#pragma unroll
            for (int off = 1; off < 16; off <<= 1)
                sum += __shfl_xor(sum, off);
            lrun[r] = lrun[r] * scl + sum;
#pragma unroll
            for (int j = 0; j < 8; ++j) oacc[j][r] *= scl;
        }

        // P (C-layout) -> LDS so it can be re-read as A-fragments.
        // Intra-wave LDS ops are in-order, no barrier needed.
#pragma unroll
        for (int nt = 0; nt < 4; ++nt)
#pragma unroll
            for (int r = 0; r < 8; ++r)
                Pw[(r + 8 * half) * LDP + nt * 16 + nn] = f2bf(sacc[nt][r]);

        // O += P * V   (16x128, 16 WMMAs)
#pragma unroll
        for (int f2 = 0; f2 < 2; ++f2) {
            bf16x16 pa = frag_ld(Pw + f2 * 32, LDP);
#pragma unroll
            for (int j = 0; j < 8; ++j)
                oacc[j] = __builtin_amdgcn_wmma_f32_16x16x32_bf16(
                    false, pa, false,
                    frag_ld(&Vb[(j * 16) * LDV + f2 * 32], LDV),
                    (short)0, oacc[j], false, false);
        }
        __syncthreads();                        // protect stage kt&1 before reuse
    }

    // normalize and write ctx [B,S,D] bf16
#pragma unroll
    for (int j = 0; j < 8; ++j)
#pragma unroll
        for (int r = 0; r < 8; ++r) {
            int sg = qw + r + 8 * half;
            float v = oacc[j][r] / lrun[r];
            ctx[((size_t)b * S + sg) * D + h * Dh + j * 16 + nn] = f2bf(v);
        }
}

// ---------------------------------------------------------------------------
// Host orchestration
// ---------------------------------------------------------------------------
extern "C" void kernel_launch(void* const* d_in, const int* in_sizes, int n_in,
                              void* d_out, int out_size, void* d_ws, size_t ws_size,
                              hipStream_t stream) {
    (void)in_sizes; (void)n_in; (void)out_size; (void)ws_size;
    const float* hidden = (const float*)d_in[0];
    const int*   pos    = (const int*)d_in[1];
    const float* Wqkv   = (const float*)d_in[2];
    const float* bqkv   = (const float*)d_in[3];
    const float* Wproj  = (const float*)d_in[4];

    size_t off = 0;
    auto carve = [&](size_t bytes) {
        void* p = (char*)d_ws + off;
        off += (bytes + 255) & ~(size_t)255;
        return p;
    };
    __bf16* hbf    = (__bf16*)carve((size_t)B * S * D * 2);
    __bf16* WqkvT  = (__bf16*)carve((size_t)3 * D * D * 2);
    __bf16* WprojT = (__bf16*)carve((size_t)D * D * 2);
    __bf16* qkv    = (__bf16*)carve((size_t)B * S * 3 * D * 2);
    __bf16* Qr     = (__bf16*)carve((size_t)B * H * S * Dh * 2);
    __bf16* Kr     = (__bf16*)carve((size_t)B * H * S * Dh * 2);
    __bf16* Vtr    = (__bf16*)carve((size_t)B * H * Dh * S * 2);
    __bf16* ctx    = (__bf16*)carve((size_t)B * S * D * 2);

    const int M = B * S;                         // 4096

    // 1. precision conversion of activations and weights
    {
        size_t n = (size_t)B * S * D;
        cvt_bf16_kernel<<<(unsigned)((n + 255) / 256), 256, 0, stream>>>(hidden, hbf, n);
    }
    transpose_cvt_kernel<<<dim3(3 * D / 32, D / 32), dim3(32, 8), 0, stream>>>(Wqkv, WqkvT, D, 3 * D);
    transpose_cvt_kernel<<<dim3(D / 32, D / 32), dim3(32, 8), 0, stream>>>(Wproj, WprojT, D, D);

    // 2. QKV projection (M=4096, N=12288, K=4096), bias, bf16 out
    gemm_wmma_kernel<true, true><<<dim3(3 * D / 128, M / 128), 256, 0, stream>>>(
        hbf, WqkvT, bqkv, (void*)qkv, M, 3 * D, D);

    // 3. RoPE + repack into attention-friendly layouts
    rope_pack_kernel<<<B * S, 256, 0, stream>>>(qkv, pos, Qr, Kr, Vtr);

    // 4. causal flash attention
    flash_attn_kernel<<<dim3(S / 64, H, B), 128, 0, stream>>>(Qr, Kr, Vtr, ctx);

    // 5. output projection (M=4096, N=4096, K=4096), f32 out
    gemm_wmma_kernel<false, false><<<dim3(D / 128, M / 128), 256, 0, stream>>>(
        ctx, WprojT, nullptr, d_out, M, D, D);
}
